// key_pooling_22960895165060
// MI455X (gfx1250) — matrix operand order repaired and checked by reading, not compiled, not run on hardware
//
#include <hip/hip_runtime.h>

#define W_TOT   262144
#define L_DIM   512
#define G_DIM   128
#define ROWS_PB 128
#define NBLK    (W_TOT / ROWS_PB)   /* 2048 */
#define KC      128                 /* K chunk staged in LDS */
#define KCP     136                 /* padded K stride (f16 elems), 272B rows */
#define NEG_HUGE (-3.402823466e38f)

typedef __attribute__((ext_vector_type(16))) _Float16 v16h;
typedef __attribute__((ext_vector_type(8)))  _Float16 v8h;
typedef __attribute__((ext_vector_type(8)))  float    v8f;

// ---------------------------------------------------------------------------
// Kernel 1: per-128-row-block GEMM (f16 WMMA, f32 accum) + bias + block-local
// cumulative max. Writes block-cummaxed proj to out, block max to bmax.
// ---------------------------------------------------------------------------
__global__ __launch_bounds__(256) void gemm_blockscan_kernel(
    const float* __restrict__ A, const float* __restrict__ W1,
    const float* __restrict__ b1, float* __restrict__ out,
    float* __restrict__ bmax)
{
  __shared__ _Float16 ldsB[G_DIM * KCP];   // W1 chunk, f16, [N][K] padded
  __shared__ float    ldsStrip[8 * G_DIM]; // per-wave strip maxima

  const int tid  = threadIdx.x;
  const int lane = tid & 31;
  const int wv   = tid >> 5;      // wave 0..7, owns rows [wv*16, wv*16+16)
  const int h    = lane >> 4;     // lane half (selects K sub-block / M half)
  const int nlo  = lane & 15;
  const int blk  = blockIdx.x;
  const int row0 = blk * ROWS_PB + wv * 16;

  v8f acc[8] = {};                // 8 N-tiles of 16x16 f32 accumulators

  const int kbA = h * 8;          // A-fragment lane K base (16-bit A layout)
  const int kbB = h * 8;          // B-fragment lane K base (32x16 B layout)
  const float* aRow = A + (size_t)(row0 + nlo) * L_DIM;

  for (int k0 = 0; k0 < L_DIM; k0 += KC) {
    // Stage W1[k0:k0+128][:] -> LDS as f16, transposed [n][k] with pad.
    #pragma unroll 4
    for (int i = 0; i < 64; ++i) {
      int idx = tid + i * 256;
      int kk  = idx >> 7;         // 0..127
      int nn  = idx & 127;        // 0..127
      ldsB[nn * KCP + kk] = (_Float16)W1[(size_t)(k0 + kk) * G_DIM + nn];
    }
    __syncthreads();

    if (k0 + KC < L_DIM)
      __builtin_prefetch(aRow + k0 + KC, 0, 1);  // global_prefetch_b8

    #pragma unroll
    for (int ks = 0; ks < KC; ks += 32) {
      // Build 16x32 f16 A fragment: lane half h holds K = kbA+{0..7,16..23}.
      const float* ap = aRow + k0 + ks + kbA;
      float4 x0 = *(const float4*)(ap);
      float4 x1 = *(const float4*)(ap + 4);
      float4 y0 = *(const float4*)(ap + 16);
      float4 y1 = *(const float4*)(ap + 20);
      v16h afrag;
      afrag[0]  = (_Float16)x0.x; afrag[1]  = (_Float16)x0.y;
      afrag[2]  = (_Float16)x0.z; afrag[3]  = (_Float16)x0.w;
      afrag[4]  = (_Float16)x1.x; afrag[5]  = (_Float16)x1.y;
      afrag[6]  = (_Float16)x1.z; afrag[7]  = (_Float16)x1.w;
      afrag[8]  = (_Float16)y0.x; afrag[9]  = (_Float16)y0.y;
      afrag[10] = (_Float16)y0.z; afrag[11] = (_Float16)y0.w;
      afrag[12] = (_Float16)y1.x; afrag[13] = (_Float16)y1.y;
      afrag[14] = (_Float16)y1.z; afrag[15] = (_Float16)y1.w;

      #pragma unroll
      for (int t = 0; t < 8; ++t) {
        // 32x16 B fragment: lane holds column n = nlo+16t, K = kbB+{0..7,16..23}.
        const v8h* bp = (const v8h*)&ldsB[(nlo + 16 * t) * KCP + ks + kbB];
        v8h blo = bp[0], bhi = bp[2];
        v16h bfrag;
        #pragma unroll
        for (int e = 0; e < 8; ++e) { bfrag[e] = blo[e]; bfrag[8 + e] = bhi[e]; }
        acc[t] = __builtin_amdgcn_wmma_f32_16x16x32_f16(
            false, afrag, false, bfrag, (short)0, acc[t], false, false);
      }
    }
    __syncthreads();
  }

  // Bias add. C/D layout: lane(l) VGPR(r) = element (M = h*8 + r, N = nlo+16t).
  #pragma unroll
  for (int t = 0; t < 8; ++t) {
    float bb = b1[nlo + 16 * t];
    #pragma unroll
    for (int r = 0; r < 8; ++r) acc[t][r] += bb;
  }

  // --- intra-strip (16-row) cumulative max, per column ---
  #pragma unroll
  for (int t = 0; t < 8; ++t) {
    #pragma unroll
    for (int r = 1; r < 8; ++r) acc[t][r] = fmaxf(acc[t][r], acc[t][r - 1]);
  }
  // Carry low-half (rows 0..7) column max into upper half via ds_bpermute.
  #pragma unroll
  for (int t = 0; t < 8; ++t) {
    int packed = __builtin_amdgcn_ds_bpermute(nlo << 2, __float_as_int(acc[t][7]));
    float lowtop = __int_as_float(packed);
    if (h) {
      #pragma unroll
      for (int r = 0; r < 8; ++r) acc[t][r] = fmaxf(acc[t][r], lowtop);
    }
  }

  // --- cross-wave prefix within the 128-row block ---
  if (h) {
    #pragma unroll
    for (int t = 0; t < 8; ++t)
      ldsStrip[wv * G_DIM + nlo + 16 * t] = acc[t][7];  // strip total
  }
  __syncthreads();
  #pragma unroll
  for (int t = 0; t < 8; ++t) {
    float pre = NEG_HUGE;
    for (int ww = 0; ww < wv; ++ww)
      pre = fmaxf(pre, ldsStrip[ww * G_DIM + nlo + 16 * t]);
    #pragma unroll
    for (int r = 0; r < 8; ++r) acc[t][r] = fmaxf(acc[t][r], pre);
  }

  // Store block-cummaxed proj; last wave's upper half holds the block total.
  float* obase = out + (size_t)(row0 + h * 8) * G_DIM;
  #pragma unroll
  for (int t = 0; t < 8; ++t) {
    int n = nlo + 16 * t;
    #pragma unroll
    for (int r = 0; r < 8; ++r) obase[(size_t)r * G_DIM + n] = acc[t][r];
    if (wv == 7 && h == 1) bmax[blk * G_DIM + n] = acc[t][7];
  }
}

// ---------------------------------------------------------------------------
// Kernel 2: exclusive max-scan over the 2048 block maxima, seeded by g.
// ---------------------------------------------------------------------------
__global__ __launch_bounds__(128) void block_prefix_kernel(
    const float* __restrict__ bmax, const float* __restrict__ g,
    float* __restrict__ prefix)
{
  int t = threadIdx.x;             // one column per thread
  float run = g[t];
  for (int b = 0; b < NBLK; ++b) {
    prefix[b * G_DIM + t] = run;
    run = fmaxf(run, bmax[b * G_DIM + t]);
  }
}

// ---------------------------------------------------------------------------
// Kernel 3: out = max(out, prefix[block]) elementwise; last block also writes
// the trailing global_feature_new row.
// ---------------------------------------------------------------------------
__global__ __launch_bounds__(256) void apply_prefix_kernel(
    float* __restrict__ out, const float* __restrict__ prefix)
{
  __shared__ float pref[G_DIM];
  const int tid = threadIdx.x;
  const int b   = blockIdx.x;
  if (tid < G_DIM) pref[tid] = prefix[b * G_DIM + tid];
  __syncthreads();

  float4* o4 = (float4*)out + (size_t)b * (ROWS_PB * G_DIM / 4);
  #pragma unroll 4
  for (int i = 0; i < 16; ++i) {
    int id = tid + i * 256;        // float4 index within block tile
    int c  = (id * 4) & (G_DIM - 1);
    float4 v = o4[id];
    v.x = fmaxf(v.x, pref[c + 0]);
    v.y = fmaxf(v.y, pref[c + 1]);
    v.z = fmaxf(v.z, pref[c + 2]);
    v.w = fmaxf(v.w, pref[c + 3]);
    o4[id] = v;
    if (b == NBLK - 1 && id >= (ROWS_PB * G_DIM / 4 - G_DIM / 4)) {
      float* tail = out + (size_t)W_TOT * G_DIM;   // global_feature_new
      tail[c + 0] = v.x; tail[c + 1] = v.y;
      tail[c + 2] = v.z; tail[c + 3] = v.w;
    }
  }
}

extern "C" void kernel_launch(void* const* d_in, const int* in_sizes, int n_in,
                              void* d_out, int out_size, void* d_ws, size_t ws_size,
                              hipStream_t stream) {
  (void)in_sizes; (void)n_in; (void)out_size; (void)ws_size;
  const float* local_features = (const float*)d_in[0];   // [W, L]
  const float* global_feature = (const float*)d_in[1];   // [1, G]
  const float* W1             = (const float*)d_in[2];   // [L, G]
  const float* b1             = (const float*)d_in[3];   // [G]
  float* out = (float*)d_out;                            // [W*G] ++ [G]

  float* bmax   = (float*)d_ws;                 // [NBLK * G]
  float* prefix = bmax + (size_t)NBLK * G_DIM;  // [NBLK * G]

  gemm_blockscan_kernel<<<NBLK, 256, 0, stream>>>(
      local_features, W1, b1, out, bmax);
  block_prefix_kernel<<<1, 128, 0, stream>>>(bmax, global_feature, prefix);
  apply_prefix_kernel<<<NBLK, 256, 0, stream>>>(out, prefix);
}
